// Net_86440511799409
// MI455X (gfx1250) — matrix operand order repaired and checked by reading, not compiled, not run on hardware
//
#include <hip/hip_runtime.h>
#include <hip/hip_bf16.h>

typedef __attribute__((ext_vector_type(16))) _Float16 v16h;
typedef __attribute__((ext_vector_type(8)))  float    v8f;

#define WMMA_F16(a, b, c) \
  __builtin_amdgcn_wmma_f32_16x16x32_f16(false, (a), false, (b), (short)0, (c), false, false)

// ---------------------------------------------------------------------------
// WMMA fragment-order LDS addressing (wave32 layouts per CDNA5 ISA 7.12.2).
// Tiles are stored so each lane's 16 halfs are CONTIGUOUS (512 halfs / tile):
// fragment reads become 2x ds_load_b128 instead of 16x ds_load_u16 + moves.
// ---------------------------------------------------------------------------

// A tile: 16(M) x 32(K).  element(m,k) lives at lane = m + 16*((k>>3)&1),
//                                          e    = (k&7) + 8*(k>>4)
__device__ __forceinline__ int afrag_idx(int m, int k) {
  int lane = (m & 15) + (((k >> 3) & 1) << 4);
  int e = (k & 7) + ((k >> 4) << 3);
  return lane * 16 + e;
}
// B tile: 32(K) x 16(N).  element(k,n) lives at lane = n + 16*((k>>3)&1),
//                                          e    = (k&7) + 8*(k>>4)
__device__ __forceinline__ int bfrag_idx(int k, int n) {
  int lane = (n & 15) + (((k >> 3) & 1) << 4);
  int e = (k & 7) + ((k >> 4) << 3);
  return lane * 16 + e;
}
__device__ __forceinline__ v16h frag_load(const _Float16* base, int lane) {
  return *(const v16h*)(base + lane * 16);
}
__device__ __forceinline__ v8f v8f_zero() {
  v8f z;
#pragma unroll
  for (int r = 0; r < 8; ++r) z[r] = 0.0f;
  return z;
}
__device__ __forceinline__ v16h v16h_zero() {
  v16h z;
#pragma unroll
  for (int e = 0; e < 16; ++e) z[e] = (_Float16)0.0f;
  return z;
}

// ---------------------------------------------------------------------------
// proj: P[N x HD] = X^T @ W     X channel-major [C][N], W [C][HD], C % 32 == 0
// block = 128 thr (4 waves), tile 64 rows x HD cols; grid = (ceil(N/64), batch)
// ---------------------------------------------------------------------------
template <int HD>
__global__ __launch_bounds__(128) void proj_kernel(const float* __restrict__ X,
                                                   const float* __restrict__ W,
                                                   float* __restrict__ P, int C, int N) {
  __shared__ __align__(32) _Float16 As[4 * 512];            // 64x32 A tiles (one k-step)
  __shared__ __align__(32) _Float16 Ws[(HD / 16) * 512];    // 32xHD B tiles (one k-step)
  int b = blockIdx.y;
  X += (size_t)b * C * N;
  P += (size_t)b * N * HD;
  int m0 = blockIdx.x * 64;
  int tid = threadIdx.x, wave = tid >> 5, lane = tid & 31;

  v8f acc[HD / 16];
#pragma unroll
  for (int t = 0; t < HD / 16; ++t) acc[t] = v8f_zero();

  for (int k0 = 0; k0 < C; k0 += 32) {
    for (int i = tid; i < 64 * 32; i += 128) {
      int m = i & 63, k = i >> 6;                       // coalesced over global N dim
      int gm = m0 + m;
      _Float16 v = (gm < N) ? (_Float16)X[(size_t)(k0 + k) * N + gm] : (_Float16)0.0f;
      As[(m >> 4) * 512 + afrag_idx(m, k)] = v;
    }
    for (int i = tid; i < 32 * HD; i += 128) {
      int k = i / HD, n = i % HD;
      Ws[(n >> 4) * 512 + bfrag_idx(k, n)] = (_Float16)W[(size_t)(k0 + k) * HD + n];
    }
    __syncthreads();
    v16h a = frag_load(&As[wave * 512], lane);
#pragma unroll
    for (int t = 0; t < HD / 16; ++t) {
      v16h bf = frag_load(&Ws[t * 512], lane);
      acc[t] = WMMA_F16(a, bf, acc[t]);
    }
    __syncthreads();
  }
  int n = lane & 15, rb = (lane & 16) ? 8 : 0;
  int baseRow = m0 + wave * 16 + rb;
#pragma unroll
  for (int t = 0; t < HD / 16; ++t)
#pragma unroll
    for (int r = 0; r < 8; ++r) {
      int row = baseRow + r;
      if (row < N) P[(size_t)row * HD + t * 16 + n] = acc[t][r];   // N2=196 tail guard
    }
}

// ---------------------------------------------------------------------------
// attention pass 1: global max of logits (x1 @ x2^T), deterministic partials
// ---------------------------------------------------------------------------
template <int HD>
__global__ __launch_bounds__(128) void attn_max_kernel(const float* __restrict__ X1,
                                                       const float* __restrict__ X2,
                                                       float* __restrict__ blockMax,
                                                       int N1, int N2, int nBlk) {
  __shared__ __align__(32) _Float16 x1F[4 * (HD / 32) * 512];   // A tiles: [sub][kstep]
  __shared__ __align__(32) _Float16 x2btF[2 * (HD / 32) * 512]; // B=x2^T: [coltile][kstep]
  __shared__ float red[128];
  int b = blockIdx.y;
  X1 += (size_t)b * N1 * HD;
  X2 += (size_t)b * N2 * HD;
  int m0 = blockIdx.x * 64;
  int tid = threadIdx.x, wave = tid >> 5, lane = tid & 31;

  for (int i = tid; i < 64 * HD; i += 128) {
    int m = i / HD, h = i % HD;
    _Float16 v = (m0 + m < N1) ? (_Float16)X1[(size_t)(m0 + m) * HD + h] : (_Float16)0.0f;
    x1F[((m >> 4) * (HD / 32) + (h >> 5)) * 512 + afrag_idx(m, h & 31)] = v;
  }
  __syncthreads();

  bool rowok = (m0 + wave * 16) < N1;   // N1 is a multiple of 16
  float vmax = -3.0e38f;
  int n = lane & 15;

  for (int n0 = 0; n0 < N2; n0 += 32) {
    for (int i = tid; i < 32 * HD; i += 128) {
      int m = i / HD, h = i % HD;
      _Float16 v = (n0 + m < N2) ? (_Float16)X2[(size_t)(n0 + m) * HD + h] : (_Float16)0.0f;
      x2btF[((m >> 4) * (HD / 32) + (h >> 5)) * 512 + bfrag_idx(h & 31, m)] = v;
    }
    __syncthreads();
    v8f c0 = v8f_zero(), c1 = v8f_zero();
#pragma unroll
    for (int ks = 0; ks < HD / 32; ++ks) {
      v16h a = frag_load(&x1F[(wave * (HD / 32) + ks) * 512], lane);
      v16h b0 = frag_load(&x2btF[ks * 512], lane);
      v16h b1 = frag_load(&x2btF[((HD / 32) + ks) * 512], lane);
      c0 = WMMA_F16(a, b0, c0);
      c1 = WMMA_F16(a, b1, c1);
    }
    if (rowok) {
      bool ok0 = (n0 + n) < N2, ok1 = (n0 + 16 + n) < N2;
#pragma unroll
      for (int r = 0; r < 8; ++r) {
        if (ok0) vmax = fmaxf(vmax, c0[r]);
        if (ok1) vmax = fmaxf(vmax, c1[r]);
      }
    }
    __syncthreads();
  }

  red[tid] = vmax;
  __syncthreads();
  for (int s = 64; s > 0; s >>= 1) {
    if (tid < s) red[tid] = fmaxf(red[tid], red[tid + s]);
    __syncthreads();
  }
  if (tid == 0) blockMax[b * nBlk + blockIdx.x] = red[0];
}

// ---------------------------------------------------------------------------
// attention pass 2: numer[n] = sum_m exp(a_nm - gmax) * x2_m ; partial Z sums
// flash-style recompute; exp tile written in A-fragment order for 2nd WMMA
// ---------------------------------------------------------------------------
template <int HD>
__global__ __launch_bounds__(128) void attn_numer_kernel(const float* __restrict__ X1,
                                                         const float* __restrict__ X2,
                                                         const float* __restrict__ gmax,
                                                         float* __restrict__ numer,
                                                         float* __restrict__ blockSum,
                                                         int N1, int N2, int nBlk) {
  __shared__ __align__(32) _Float16 x1F[4 * (HD / 32) * 512];
  __shared__ __align__(32) _Float16 x2btF[2 * (HD / 32) * 512]; // B = x2^T (logits)
  __shared__ __align__(32) _Float16 x2bF[(HD / 16) * 512];      // B = x2   (numer)
  __shared__ __align__(32) _Float16 esF[4 * 512];               // per-wave exp tile (A order)
  __shared__ float red[128];
  int b = blockIdx.y;
  X1 += (size_t)b * N1 * HD;
  X2 += (size_t)b * N2 * HD;
  numer += (size_t)b * N1 * HD;
  int m0 = blockIdx.x * 64;
  int tid = threadIdx.x, wave = tid >> 5, lane = tid & 31;

  for (int i = tid; i < 64 * HD; i += 128) {
    int m = i / HD, h = i % HD;
    _Float16 v = (m0 + m < N1) ? (_Float16)X1[(size_t)(m0 + m) * HD + h] : (_Float16)0.0f;
    x1F[((m >> 4) * (HD / 32) + (h >> 5)) * 512 + afrag_idx(m, h & 31)] = v;
  }
  __syncthreads();

  bool rowok = (m0 + wave * 16) < N1;
  float mx = gmax[b];
  float zsum = 0.0f;
  int n = lane & 15, rb = (lane & 16) ? 8 : 0;

  v8f nacc[HD / 16];
#pragma unroll
  for (int t = 0; t < HD / 16; ++t) nacc[t] = v8f_zero();

  for (int n0 = 0; n0 < N2; n0 += 32) {
    for (int i = tid; i < 32 * HD; i += 128) {
      int m = i / HD, h = i % HD;
      _Float16 v = (n0 + m < N2) ? (_Float16)X2[(size_t)(n0 + m) * HD + h] : (_Float16)0.0f;
      x2btF[((m >> 4) * (HD / 32) + (h >> 5)) * 512 + bfrag_idx(h & 31, m)] = v;
      x2bF[(h >> 4) * 512 + bfrag_idx(m, h)] = v;
    }
    __syncthreads();

    v8f c0 = v8f_zero(), c1 = v8f_zero();
#pragma unroll
    for (int ks = 0; ks < HD / 32; ++ks) {
      v16h a = frag_load(&x1F[(wave * (HD / 32) + ks) * 512], lane);
      v16h b0 = frag_load(&x2btF[ks * 512], lane);
      v16h b1 = frag_load(&x2btF[((HD / 32) + ks) * 512], lane);
      c0 = WMMA_F16(a, b0, c0);
      c1 = WMMA_F16(a, b1, c1);
    }

    bool ok0 = (n0 + n) < N2, ok1 = (n0 + 16 + n) < N2;
#pragma unroll
    for (int r = 0; r < 8; ++r) {
      float e0 = (rowok && ok0) ? __expf(c0[r] - mx) : 0.0f;
      float e1 = (rowok && ok1) ? __expf(c1[r] - mx) : 0.0f;
      zsum += e0 + e1;
      esF[wave * 512 + afrag_idx(rb + r, n)] = (_Float16)e0;       // col k = n
      esF[wave * 512 + afrag_idx(rb + r, 16 + n)] = (_Float16)e1;  // col k = 16+n
    }
    // same-wave LDS ops are kept in order by hardware
    v16h ae = frag_load(&esF[wave * 512], lane);
#pragma unroll
    for (int t = 0; t < HD / 16; ++t) {
      v16h bf = frag_load(&x2bF[t * 512], lane);
      nacc[t] = WMMA_F16(ae, bf, nacc[t]);
    }
    __syncthreads();
  }

  if (rowok) {
    int baseRow = m0 + wave * 16 + rb;
#pragma unroll
    for (int t = 0; t < HD / 16; ++t)
#pragma unroll
      for (int r = 0; r < 8; ++r)
        numer[(size_t)(baseRow + r) * HD + t * 16 + n] = nacc[t][r];
  }

  red[tid] = zsum;
  __syncthreads();
  for (int s = 64; s > 0; s >>= 1) {
    if (tid < s) red[tid] += red[tid + s];
    __syncthreads();
  }
  if (tid == 0) blockSum[b * nBlk + blockIdx.x] = red[0];
}

// ---------------------------------------------------------------------------
// deterministic partial reductions (no float atomics -> bitwise reproducible)
// ---------------------------------------------------------------------------
__global__ void reduce_max_kernel(const float* __restrict__ part, float* __restrict__ out, int n) {
  __shared__ float red[256];
  int b = blockIdx.x;
  float v = -3.0e38f;
  for (int i = threadIdx.x; i < n; i += 256) v = fmaxf(v, part[b * n + i]);
  red[threadIdx.x] = v;
  __syncthreads();
  for (int s = 128; s > 0; s >>= 1) {
    if (threadIdx.x < s) red[threadIdx.x] = fmaxf(red[threadIdx.x], red[threadIdx.x + s]);
    __syncthreads();
  }
  if (threadIdx.x == 0) out[b] = red[0];
}

__global__ void reduce_sum_kernel(const float* __restrict__ part, float* __restrict__ out, int n) {
  __shared__ float red[256];
  int b = blockIdx.x;
  float v = 0.0f;
  for (int i = threadIdx.x; i < n; i += 256) v += part[b * n + i];
  red[threadIdx.x] = v;
  __syncthreads();
  for (int s = 128; s > 0; s >>= 1) {
    if (threadIdx.x < s) red[threadIdx.x] += red[threadIdx.x + s];
    __syncthreads();
  }
  if (threadIdx.x == 0) out[b] = red[0];
}

// out[b][h][n] = numer[b][n][h] / Z[b] + x1[b][n][h]   (transpose to channel-major)
__global__ void csa_finalize_kernel(const float* __restrict__ numer, const float* __restrict__ x1,
                                    const float* __restrict__ Z, float* __restrict__ out,
                                    int N1, int HD) {
  int b = blockIdx.y;
  float invZ = 1.0f / Z[b];
  size_t tot = (size_t)N1 * HD;
  numer += (size_t)b * tot;
  x1 += (size_t)b * tot;
  out += (size_t)b * tot;
  for (size_t idx = blockIdx.x * 256 + threadIdx.x; idx < tot; idx += (size_t)gridDim.x * 256) {
    int h = (int)(idx / N1), n = (int)(idx % N1);
    size_t src = (size_t)n * HD + h;
    out[idx] = numer[src] * invZ + x1[src];
  }
}

// ---------------------------------------------------------------------------
// MSR via WMMA: per-pixel MLP (fc1 + relu6 + fc2) as 64-pixel GEMM tiles.
// Bilinear x2 upsample of `coarse` fused into input-tile staging; channel dims
// zero-padded to WMMA shapes (exact: padded weights/biases are zero).
// ---------------------------------------------------------------------------
template <int CIN, int CHID, int COUT>
__global__ __launch_bounds__(128) void msr_wmma_kernel(
    const float* __restrict__ fine, const float* __restrict__ coarse,
    const float* __restrict__ W1g, const float* __restrict__ B1g,
    const float* __restrict__ W2g, const float* __restrict__ B2g,
    float* __restrict__ out, int H, int W, int batch) {
  constexpr int CINP = (CIN + 31) & ~31;   // fc1 K (pad to 32)
  constexpr int CHN = (CHID + 15) & ~15;   // fc1 N (pad to 16)
  constexpr int CHK = (CHID + 31) & ~31;   // fc2 K (pad to 32)
  constexpr int COP = (COUT + 15) & ~15;   // fc2 N (pad to 16)
  __shared__ __align__(32) _Float16 insF[4 * (CINP / 32) * 512];
  __shared__ __align__(32) _Float16 w1F[(CINP / 32) * (CHN / 16) * 512];
  __shared__ __align__(32) _Float16 w2F[(CHK / 32) * (COP / 16) * 512];
  __shared__ __align__(32) _Float16 hidF[4 * (CHK / 32) * 512];
  __shared__ float b1s[CHN];
  __shared__ float b2s[COP];
  int HW = H * W, Hc = H >> 1, Wc = W >> 1;
  int tid = threadIdx.x, wave = tid >> 5, lane = tid & 31;
  int n = lane & 15, rb = (lane & 16) ? 8 : 0;

  // stage (zero-padded) weights once per block
  for (int i = tid; i < CINP * CHN; i += 128) {
    int c = i / CHN, j = i % CHN;
    float v = (c < CIN && j < CHID) ? W1g[c * CHID + j] : 0.0f;
    w1F[((c >> 5) * (CHN / 16) + (j >> 4)) * 512 + bfrag_idx(c & 31, j)] = (_Float16)v;
  }
  for (int i = tid; i < CHK * COP; i += 128) {
    int c = i / COP, j = i % COP;
    float v = (c < CHID && j < COUT) ? W2g[c * COUT + j] : 0.0f;
    w2F[((c >> 5) * (COP / 16) + (j >> 4)) * 512 + bfrag_idx(c & 31, j)] = (_Float16)v;
  }
  for (int i = tid; i < CHN; i += 128) b1s[i] = (i < CHID) ? B1g[i] : 0.0f;
  for (int i = tid; i < COP; i += 128) b2s[i] = (i < COUT) ? B2g[i] : 0.0f;
  __syncthreads();

  int totalTiles = (batch * HW) >> 6;   // HW % 64 == 0 at every scale
  for (int T = blockIdx.x; T < totalTiles; T += gridDim.x) {
    int p0 = T << 6;
    int b = p0 / HW;
    int pb = p0 - b * HW;
    // stage 64 x CINP input tile (fused upsample), in A-fragment order
    for (int i = tid; i < 64 * CINP; i += 128) {
      int m = i / CINP, c = i % CINP;
      float v = 0.0f;
      if (c < CIN) {
        int p = pb + m;
        if (fine) v = fine[((size_t)b * CIN + c) * HW + p];
        if (coarse) {
          int h = p / W, w = p % W;
          int hi = h >> 1, wi = w >> 1;
          int i1 = (h & 1) ? min(hi + 1, Hc - 1) : max(hi - 1, 0);
          int j1 = (w & 1) ? min(wi + 1, Wc - 1) : max(wi - 1, 0);
          const float* cp = coarse + ((size_t)b * CIN + c) * Hc * Wc;
          v += 0.5625f * cp[hi * Wc + wi] + 0.1875f * cp[hi * Wc + j1] +
               0.1875f * cp[i1 * Wc + wi] + 0.0625f * cp[i1 * Wc + j1];
        }
      }
      insF[((m >> 4) * (CINP / 32) + (c >> 5)) * 512 + afrag_idx(m, c & 31)] = (_Float16)v;
    }
    __syncthreads();

    // fc1
    v8f acc[CHN / 16];
#pragma unroll
    for (int t = 0; t < CHN / 16; ++t) acc[t] = v8f_zero();
#pragma unroll
    for (int ks = 0; ks < CINP / 32; ++ks) {
      v16h a = frag_load(&insF[(wave * (CINP / 32) + ks) * 512], lane);
#pragma unroll
      for (int t = 0; t < CHN / 16; ++t) {
        v16h bf = frag_load(&w1F[(ks * (CHN / 16) + t) * 512], lane);
        acc[t] = WMMA_F16(a, bf, acc[t]);
      }
    }
    // bias + relu6 -> hidF (A-fragment order, zero-filled padding first)
#pragma unroll
    for (int ks = 0; ks < CHK / 32; ++ks)
      *(v16h*)&hidF[(wave * (CHK / 32) + ks) * 512 + lane * 16] = v16h_zero();
#pragma unroll
    for (int t = 0; t < CHN / 16; ++t) {
      int j = t * 16 + n;
      float bias = b1s[j];
#pragma unroll
      for (int r = 0; r < 8; ++r) {
        float hv = fminf(fmaxf(acc[t][r] + bias, 0.0f), 6.0f);
        hidF[(wave * (CHK / 32) + (j >> 5)) * 512 + afrag_idx(rb + r, j & 31)] = (_Float16)hv;
      }
    }
    // fc2 (same-wave LDS in-order: safe to reload immediately)
    v8f acc2[COP / 16];
#pragma unroll
    for (int t = 0; t < COP / 16; ++t) acc2[t] = v8f_zero();
#pragma unroll
    for (int ks = 0; ks < CHK / 32; ++ks) {
      v16h a = frag_load(&hidF[(wave * (CHK / 32) + ks) * 512], lane);
#pragma unroll
      for (int t = 0; t < COP / 16; ++t) {
        v16h bf = frag_load(&w2F[(ks * (COP / 16) + t) * 512], lane);
        acc2[t] = WMMA_F16(a, bf, acc2[t]);
      }
    }
    // store valid outputs
    int prow = pb + wave * 16 + rb;
#pragma unroll
    for (int t = 0; t < COP / 16; ++t) {
      int o = t * 16 + n;
      if (o < COUT) {
        float bias = b2s[o];
#pragma unroll
        for (int r = 0; r < 8; ++r)
          out[((size_t)b * COUT + o) * HW + prow + r] = acc2[t][r] + bias;
      }
    }
    __syncthreads();
  }
}

// ---------------------------------------------------------------------------
// host orchestration
// ---------------------------------------------------------------------------
static inline int ceil_div_i(int a, int b) { return (a + b - 1) / b; }

template <int HD>
static void run_csa(const float* hr, int C1, int N1, const float* lr, int C2, int N2,
                    const float* w1, const float* w2, float* x1buf, float* x2buf, float* numer,
                    float* pmax, float* psum, float* gmax, float* zsum, float* out,
                    hipStream_t stream) {
  dim3 blk(128);
  proj_kernel<HD><<<dim3(ceil_div_i(N1, 64), 2), blk, 0, stream>>>(hr, w1, x1buf, C1, N1);
  proj_kernel<HD><<<dim3(ceil_div_i(N2, 64), 2), blk, 0, stream>>>(lr, w2, x2buf, C2, N2);
  int nBlk = ceil_div_i(N1, 64);
  attn_max_kernel<HD><<<dim3(nBlk, 2), blk, 0, stream>>>(x1buf, x2buf, pmax, N1, N2, nBlk);
  reduce_max_kernel<<<2, 256, 0, stream>>>(pmax, gmax, nBlk);
  attn_numer_kernel<HD><<<dim3(nBlk, 2), blk, 0, stream>>>(x1buf, x2buf, gmax, numer, psum, N1,
                                                           N2, nBlk);
  reduce_sum_kernel<<<2, 256, 0, stream>>>(psum, zsum, nBlk);
  int tot = N1 * HD;
  csa_finalize_kernel<<<dim3(ceil_div_i(tot, 256), 2), 256, 0, stream>>>(numer, x1buf, zsum, out,
                                                                         N1, HD);
}

extern "C" void kernel_launch(void* const* d_in, const int* in_sizes, int n_in, void* d_out,
                              int out_size, void* d_ws, size_t ws_size, hipStream_t stream) {
  (void)in_sizes; (void)n_in; (void)out_size; (void)ws_size;
  const float* f0 = (const float*)d_in[0];   // [2,160,14,14]
  const float* f1 = (const float*)d_in[1];   // [2,128,28,28]
  const float* f2 = (const float*)d_in[2];   // [2,64,56,56]
  const float* f3 = (const float*)d_in[3];   // [2,32,112,112]
  const float* cas1_w1 = (const float*)d_in[4];
  const float* cas1_w2 = (const float*)d_in[5];
  const float* cas2_w1 = (const float*)d_in[6];
  const float* cas2_w2 = (const float*)d_in[7];
  const float* cas3_w1 = (const float*)d_in[8];
  const float* cas3_w2 = (const float*)d_in[9];

  float* ws = (float*)d_ws;
  size_t off = 0;
  auto alloc = [&](size_t cnt) {
    float* p = ws + off;
    off += (cnt + 63) & ~(size_t)63;
    return p;
  };
  float* csa1_out = alloc(2 * 64 * 784);     // [2,64,28,28]
  float* csa2_out = alloc(2 * 64 * 3136);    // [2,64,56,56]
  float* csa3_out = alloc(2 * 32 * 12544);   // [2,32,112,112]
  float* y1 = alloc(2 * 32 * 3136);          // [2,32,56,56]
  float* y2 = alloc(2 * 16 * 12544);         // [2,16,112,112]
  float* y3 = alloc(2 * 8 * 50176);          // [2,8,224,224]
  float* x1buf = alloc(2 * 12544 * 64);
  float* x2buf = alloc(2 * 3136 * 64);
  float* numer = alloc(2 * 12544 * 64);
  float* pmax = alloc(512);
  float* psum = alloc(512);
  float* gmax = alloc(64);
  float* zsum = alloc(64);

  // CSA1: hr=f1 (C=128, N=784), lr=f0 (C=160, N=196), H=64
  run_csa<64>(f1, 128, 784, f0, 160, 196, cas1_w1, cas1_w2, x1buf, x2buf, numer, pmax, psum,
              gmax, zsum, csa1_out, stream);
  // CSA2: hr=f2 (64, 3136), lr=csa1_out (64, 784), H=64
  run_csa<64>(f2, 64, 3136, csa1_out, 64, 784, cas2_w1, cas2_w2, x1buf, x2buf, numer, pmax, psum,
              gmax, zsum, csa2_out, stream);
  // CSA3: hr=f3 (32, 12544), lr=csa2_out (64, 3136), H=32
  run_csa<32>(f3, 32, 12544, csa2_out, 64, 3136, cas3_w1, cas3_w2, x1buf, x2buf, numer, pmax,
              psum, gmax, zsum, csa3_out, stream);

  // y1 = msr1(csa2_out + up2(csa1_out)) : 64->64->32 @ 56x56
  msr_wmma_kernel<64, 64, 32><<<(2 * 3136) / 64, 128, 0, stream>>>(
      csa2_out, csa1_out, (const float*)d_in[10], (const float*)d_in[11], (const float*)d_in[12],
      (const float*)d_in[13], y1, 56, 56, 2);
  // y2 = msr2(csa3_out + up2(y1)) : 32->32->16 @ 112x112
  msr_wmma_kernel<32, 32, 16><<<(2 * 12544) / 64, 128, 0, stream>>>(
      csa3_out, y1, (const float*)d_in[14], (const float*)d_in[15], (const float*)d_in[16],
      (const float*)d_in[17], y2, 112, 112, 2);
  // y3 = msr3(up2(y2)) : 16->16->8 @ 224x224
  msr_wmma_kernel<16, 16, 8><<<(2 * 50176) / 64, 128, 0, stream>>>(
      nullptr, y2, (const float*)d_in[18], (const float*)d_in[19], (const float*)d_in[20],
      (const float*)d_in[21], y3, 224, 224, 2);
  // y4 = msr4(up2(y3)) : 8->2->1 @ 448x448 -> d_out
  msr_wmma_kernel<8, 2, 1><<<(2 * 200704) / 64, 128, 0, stream>>>(
      nullptr, y3, (const float*)d_in[22], (const float*)d_in[23], (const float*)d_in[24],
      (const float*)d_in[25], (float*)d_out, 448, 448, 2);
}
